// KGAT_10445360464163
// MI455X (gfx1250) — matrix operand order repaired and checked by reading, not compiled, not run on hardware
//
#include <hip/hip_runtime.h>

typedef __attribute__((ext_vector_type(16))) _Float16 v16h;
typedef __attribute__((ext_vector_type(8)))  float    v8f;
typedef __attribute__((ext_vector_type(4)))  float    v4f;

#define KG_BATCH   16384
#define EMBED_DIM  128
#define N_REL      24
#define MAX_TILES  (KG_BATCH/16 + N_REL)   // upper bound on sum of ceil(cnt/16)
// packed-B layout: [rel][nt(8)][kt(4)][lane(32)][16 halves]
#define PACK_GROUPS (N_REL * 8 * 4 * 32)   // 24576 groups of 16 halves
#define RM_ROWS    (KG_BATCH + 1)          // +1 dummy row for padded-tile stores

// ---------------------------------------------------------------- init/count
__global__ void __launch_bounds__(32) k_init(int* counts) {
    if (threadIdx.x < N_REL) counts[threadIdx.x] = 0;
}

__global__ void __launch_bounds__(256) k_count(const int* __restrict__ r, int* counts) {
    int b = blockIdx.x * 256 + threadIdx.x;
    if (b < KG_BATCH) atomicAdd(&counts[r[b]], 1);
}

__global__ void k_scan(const int* __restrict__ counts, int* offsets, int* cursors) {
    if (threadIdx.x == 0) {
        int acc = 0;
        for (int i = 0; i < N_REL; ++i) { offsets[i] = acc; cursors[i] = acc; acc += counts[i]; }
        offsets[N_REL] = acc;
    }
}

__global__ void __launch_bounds__(256) k_scatter(const int* __restrict__ r, int* cursors,
                                                 int* __restrict__ perm,
                                                 int* __restrict__ iperm) {
    int b = blockIdx.x * 256 + threadIdx.x;
    if (b < KG_BATCH) {
        int p = atomicAdd(&cursors[r[b]], 1);
        perm[p]  = b;   // sorted position -> original row
        iperm[b] = p;   // original row -> sorted position
    }
}

// ---------------------------------------------------------------- pack trans_M
// Pre-shuffle trans_M (f32, [rel][K=128][N=128]) into WMMA B-fragment layout
// (f16), so k_gemm can load each 32x16 B fragment as one 32B/lane load.
// Fragment element j of lane L:  K = kt*32 + (L>=16)*8 + j + (j>=8)*8,  N = nt*16 + (L&15)
__global__ void __launch_bounds__(256) k_pack(const float* __restrict__ transM,
                                              _Float16* __restrict__ wb) {
    int gid = blockIdx.x * 256 + threadIdx.x;
    if (gid >= PACK_GROUPS) return;
    int lane = gid & 31;
    int kt   = (gid >> 5) & 3;
    int nt   = (gid >> 7) & 7;
    int rel  = gid >> 10;
    int K0   = kt * 32 + (lane >> 4) * 8;
    int N    = nt * 16 + (lane & 15);
    const float* src = transM + (size_t)rel * (EMBED_DIM * EMBED_DIM)
                              + (size_t)K0 * EMBED_DIM + N;
    v16h o;
#pragma unroll
    for (int j = 0; j < 16; ++j) o[j] = (_Float16)src[(size_t)(j + (j >= 8 ? 8 : 0)) * EMBED_DIM];
    *(v16h*)(wb + (size_t)gid * 16) = o;
}

// ------------------------------------------------------------ WMMA GEMM pass
// A fragment (16x32 f16, wave32): lane<16 -> M=lane, K = {b..b+7, b+16..b+23}
//                                 lane>=16 -> same M, K window shifted by +8
__device__ inline v16h load_a_frag(const float* __restrict__ p) {
    // p points at embed_row + kt*32 + (lane>=16 ? 8 : 0); 32B aligned
    const v4f* q = (const v4f*)p;
    v4f x0 = q[0], x1 = q[1];   // K base..base+7
    v4f x2 = q[4], x3 = q[5];   // K base+16..base+23
    v16h a;
#pragma unroll
    for (int i = 0; i < 4; ++i) {
        a[i]      = (_Float16)x0[i];
        a[i + 4]  = (_Float16)x1[i];
        a[i + 8]  = (_Float16)x2[i];
        a[i + 12] = (_Float16)x3[i];
    }
    return a;
}

__global__ void __launch_bounds__(32) k_gemm(
    const float*    __restrict__ embed,   // [300000,128] f32
    const _Float16* __restrict__ wb,      // packed B fragments (f16)
    const int*      __restrict__ perm,
    const int*      __restrict__ offsets, // [25]
    const int*      __restrict__ h,
    const int*      __restrict__ pos_t,
    const int*      __restrict__ neg_t,
    float* __restrict__ rmh, float* __restrict__ rmp, float* __restrict__ rmn)
{
    // map blockIdx.x -> (relation, tile-within-segment)
    int t = blockIdx.x;
    int rel, seg_begin = 0, seg_end = 0;
    for (rel = 0; rel < N_REL; ++rel) {
        seg_begin = offsets[rel];
        seg_end   = offsets[rel + 1];
        int ntile = (seg_end - seg_begin + 15) >> 4;
        if (t < ntile) break;
        t -= ntile;
    }
    if (rel >= N_REL) return;

    const int lane  = threadIdx.x;          // 0..31
    const int mrow  = lane & 15;
    const int tbase = seg_begin + (t << 4);

    int seg  = tbase + mrow;
    int segc = (seg < seg_end) ? seg : (seg_end - 1);
    int orig = perm[segc];

    const long rowH = (long)h[orig]     * EMBED_DIM;
    const long rowP = (long)pos_t[orig] * EMBED_DIM;
    const long rowN = (long)neg_t[orig] * EMBED_DIM;
    const int  koff = (lane >> 4) * 8;

    // preload all A fragments: 4 K-tiles x 3 matrices
    v16h ah[4], ap[4], an[4];
#pragma unroll
    for (int kt = 0; kt < 4; ++kt) {
        ah[kt] = load_a_frag(embed + rowH + kt * 32 + koff);
        ap[kt] = load_a_frag(embed + rowP + kt * 32 + koff);
        an[kt] = load_a_frag(embed + rowN + kt * 32 + koff);
    }

    const int mb = (lane >> 4) * 8;         // this lane's C row base
    const int nB = lane & 15;               // C column within n-tile
    // wave-uniform: does this tile contain 16 valid (consecutive) rows?
    const bool fulltile = (tbase + 16) <= seg_end;
    // fast-path base pointers (rows tbase+mb .. tbase+mb+7, column nB)
    float* __restrict__ ph = rmh + (size_t)(tbase + mb) * EMBED_DIM + nB;
    float* __restrict__ pp = rmp + (size_t)(tbase + mb) * EMBED_DIM + nB;
    float* __restrict__ pn = rmn + (size_t)(tbase + mb) * EMBED_DIM + nB;

    // fragment base for this lane: wb[(((rel*8 + nt)*4 + kt)*32 + lane)*16]
    const _Float16* __restrict__ wrel = wb + ((size_t)rel * 8 * 4 * 32) * 16;

    for (int nt = 0; nt < 8; ++nt) {
        const _Float16* __restrict__ wnt = wrel + ((size_t)nt * 4 * 32) * 16;
        // issue all 4 B-fragment loads up front (2x b128 each)
        v16h bf0 = *(const v16h*)(wnt + ((size_t)(0 * 32 + lane)) * 16);
        v16h bf1 = *(const v16h*)(wnt + ((size_t)(1 * 32 + lane)) * 16);
        v16h bf2 = *(const v16h*)(wnt + ((size_t)(2 * 32 + lane)) * 16);
        v16h bf3 = *(const v16h*)(wnt + ((size_t)(3 * 32 + lane)) * 16);

        v8f ch = {}, cp = {}, cn = {};
        ch = __builtin_amdgcn_wmma_f32_16x16x32_f16(false, ah[0], false, bf0, (short)0, ch, false, false);
        cp = __builtin_amdgcn_wmma_f32_16x16x32_f16(false, ap[0], false, bf0, (short)0, cp, false, false);
        cn = __builtin_amdgcn_wmma_f32_16x16x32_f16(false, an[0], false, bf0, (short)0, cn, false, false);
        ch = __builtin_amdgcn_wmma_f32_16x16x32_f16(false, ah[1], false, bf1, (short)0, ch, false, false);
        cp = __builtin_amdgcn_wmma_f32_16x16x32_f16(false, ap[1], false, bf1, (short)0, cp, false, false);
        cn = __builtin_amdgcn_wmma_f32_16x16x32_f16(false, an[1], false, bf1, (short)0, cn, false, false);
        ch = __builtin_amdgcn_wmma_f32_16x16x32_f16(false, ah[2], false, bf2, (short)0, ch, false, false);
        cp = __builtin_amdgcn_wmma_f32_16x16x32_f16(false, ap[2], false, bf2, (short)0, cp, false, false);
        cn = __builtin_amdgcn_wmma_f32_16x16x32_f16(false, an[2], false, bf2, (short)0, cn, false, false);
        ch = __builtin_amdgcn_wmma_f32_16x16x32_f16(false, ah[3], false, bf3, (short)0, ch, false, false);
        cp = __builtin_amdgcn_wmma_f32_16x16x32_f16(false, ap[3], false, bf3, (short)0, cp, false, false);
        cn = __builtin_amdgcn_wmma_f32_16x16x32_f16(false, an[3], false, bf3, (short)0, cn, false, false);

        // C layout: VGPR v holds row m = mb+v, column n = nB.
        if (fulltile) {
            // rows are consecutive: constant-offset stores from one base
#pragma unroll
            for (int v = 0; v < 8; ++v) {
                ph[nt * 16 + v * EMBED_DIM] = ch[v];
                pp[nt * 16 + v * EMBED_DIM] = cp[v];
                pn[nt * 16 + v * EMBED_DIM] = cn[v];
            }
        } else {
            // boundary tile: redirect padded rows to dummy row KG_BATCH
#pragma unroll
            for (int v = 0; v < 8; ++v) {
                int segv = tbase + mb + v;
                int row  = (segv < seg_end) ? segv : KG_BATCH;
                int idx  = row * EMBED_DIM + nt * 16 + nB;
                rmh[idx] = ch[v];
                rmp[idx] = cp[v];
                rmn[idx] = cn[v];
            }
        }
    }
}

// ------------------------------------------------------------- score per row
__global__ void __launch_bounds__(32) k_score(
    const float* __restrict__ rmh, const float* __restrict__ rmp,
    const float* __restrict__ rmn, const float* __restrict__ rel_embed,
    const int* __restrict__ r, const int* __restrict__ iperm,
    float* __restrict__ terms)
{
    const int b    = blockIdx.x;
    const int lane = threadIdx.x;
    const int rel  = r[b];
    const int off  = iperm[b] * EMBED_DIM + lane * 4;   // r_mul stored in sorted order

    v4f rh = *(const v4f*)(rmh + off);
    v4f rp = *(const v4f*)(rmp + off);
    v4f rn = *(const v4f*)(rmn + off);
    v4f re = *(const v4f*)(rel_embed + rel * EMBED_DIM + lane * 4);

    v4f dp = rh + re - rp;
    v4f dn = rh + re - rn;

    float pos = 0.f, neg = 0.f, l2 = 0.f;
#pragma unroll
    for (int i = 0; i < 4; ++i) {
        pos += dp[i] * dp[i];
        neg += dn[i] * dn[i];
        l2  += rh[i] * rh[i] + re[i] * re[i] + rp[i] * rp[i] + rn[i] * rn[i];
    }
#pragma unroll
    for (int s = 16; s > 0; s >>= 1) {
        pos += __shfl_xor(pos, s, 32);
        neg += __shfl_xor(neg, s, 32);
        l2  += __shfl_xor(l2,  s, 32);
    }
    if (lane == 0) {
        float x  = neg - pos;
        float kg = fmaxf(-x, 0.0f) + log1pf(expf(-fabsf(x)));  // -log_sigmoid(x)
        terms[b] = kg + 1e-5f * 0.5f * l2;
    }
}

// ------------------------------------------------- deterministic final reduce
__global__ void __launch_bounds__(256) k_reduce(const float* __restrict__ terms,
                                                float* __restrict__ out) {
    __shared__ float s[256];
    float acc = 0.f;
    for (int i = threadIdx.x; i < KG_BATCH; i += 256) acc += terms[i];
    s[threadIdx.x] = acc;
    __syncthreads();
    for (int stride = 128; stride > 0; stride >>= 1) {
        if (threadIdx.x < stride) s[threadIdx.x] += s[threadIdx.x + stride];
        __syncthreads();
    }
    if (threadIdx.x == 0) out[0] = s[0] * (1.0f / (float)KG_BATCH);
}

// ---------------------------------------------------------------------------
extern "C" void kernel_launch(void* const* d_in, const int* in_sizes, int n_in,
                              void* d_out, int out_size, void* d_ws, size_t ws_size,
                              hipStream_t stream) {
    const float* embed     = (const float*)d_in[0];   // [300000,128]
    const float* rel_embed = (const float*)d_in[1];   // [24,128]
    const float* transM    = (const float*)d_in[2];   // [24,128,128]
    const int*   h         = (const int*)d_in[3];
    const int*   r         = (const int*)d_in[4];
    const int*   pos_t     = (const int*)d_in[5];
    const int*   neg_t     = (const int*)d_in[6];
    float*       out       = (float*)d_out;

    char* ws = (char*)d_ws;
    int*      counts  = (int*)(ws + 0);          // 24 ints
    int*      offsets = (int*)(ws + 128);        // 25 ints
    int*      cursors = (int*)(ws + 256);        // 24 ints
    int*      perm    = (int*)(ws + 384);        // 16384 ints -> ends 65920
    int*      iperm   = (int*)(ws + 66048);      // 16384 ints -> ends 131584
    _Float16* wb      = (_Float16*)(ws + 131584);// packed B: 786432 B -> ends 918016
    float*    terms   = (float*)(ws + 918528);   // 16384 f32 -> ends 984064
    float*    rmh     = (float*)(ws + 984320);   // 3 x (16384+1)*128 f32
    float*    rmp     = rmh + (size_t)RM_ROWS * EMBED_DIM;
    float*    rmn     = rmp + (size_t)RM_ROWS * EMBED_DIM;

    k_init   <<<1, 32, 0, stream>>>(counts);
    k_count  <<<KG_BATCH / 256, 256, 0, stream>>>(r, counts);
    k_scan   <<<1, 1, 0, stream>>>(counts, offsets, cursors);
    k_scatter<<<KG_BATCH / 256, 256, 0, stream>>>(r, cursors, perm, iperm);
    k_pack   <<<(PACK_GROUPS + 255) / 256, 256, 0, stream>>>(transM, wb);
    k_gemm   <<<MAX_TILES, 32, 0, stream>>>(embed, wb, perm, offsets, h, pos_t, neg_t,
                                            rmh, rmp, rmn);
    k_score  <<<KG_BATCH, 32, 0, stream>>>(rmh, rmp, rmn, rel_embed, r, iperm, terms);
    k_reduce <<<1, 256, 0, stream>>>(terms, out);
}
